// MultiHeadAttention_48670569398932
// MI455X (gfx1250) — matrix operand order repaired and checked by reading, not compiled, not run on hardware
//
#include <hip/hip_runtime.h>

#define BB 4
#define TTQ 2048
#define TSS 2048
#define DM 1024
#define HH 16
#define DHH 64
#define MM (BB*TTQ)

typedef _Float16 half_t;
typedef __attribute__((ext_vector_type(16))) _Float16 v16h;
typedef __attribute__((ext_vector_type(8)))  _Float16 v8h;
typedef __attribute__((ext_vector_type(8)))  float    v8f;

union FragH { v16h h; unsigned int u[8]; uint4 q[2]; };

__device__ __forceinline__ v8f wmma_f16(v16h a, v16h b, v8f c) {
  // D = A(16x32 f16) * B(32x16 f16) + C(16x16 f32)
  return __builtin_amdgcn_wmma_f32_16x16x32_f16(false, a, false, b, (short)0, c, false, false);
}

// gfx1250 async global->LDS copy (ASYNCcnt-tracked); per-lane LDS address.
__device__ __forceinline__ void async_load_b128(unsigned lds_off, const void* gaddr) {
  asm volatile("global_load_async_to_lds_b128 %0, %1, off"
               :: "v"(lds_off), "v"(gaddr) : "memory");
}
__device__ __forceinline__ void wait_async0() {
  asm volatile("s_wait_asynccnt 0" ::: "memory");
}

// 16-lane max-reduction butterfly on the VALU via DPP16 ROW_XMASK (0x160+mask).
// Masks 1/2/4/8 permute strictly within each 16-lane row group.
__device__ __forceinline__ float rowmax16_dpp(float v) {
  int t;
  t = __builtin_amdgcn_update_dpp(0, __float_as_int(v), 0x161, 0xf, 0xf, true);
  v = fmaxf(v, __int_as_float(t));
  t = __builtin_amdgcn_update_dpp(0, __float_as_int(v), 0x162, 0xf, 0xf, true);
  v = fmaxf(v, __int_as_float(t));
  t = __builtin_amdgcn_update_dpp(0, __float_as_int(v), 0x164, 0xf, 0xf, true);
  v = fmaxf(v, __int_as_float(t));
  t = __builtin_amdgcn_update_dpp(0, __float_as_int(v), 0x168, 0xf, 0xf, true);
  v = fmaxf(v, __int_as_float(t));
  return v;
}

// ---------------------------------------------------------------------------
// Tiled WMMA GEMM: out[M=8192, N=1024] = (A[8192,1024] * W[1024,1024] + bias)
// Block: 256 threads (8 waves, 4(m) x 2(n)); block tile 128x128, K-step 32.
// A_F16: A is f16 (ctx) else f32. HEADMAJOR: f16 out [B,H,T,DH] scaled by oscale.
// ---------------------------------------------------------------------------
template<bool A_F16, bool HEADMAJOR>
__global__ __launch_bounds__(256) void gemm_wmma_kernel(
    const void* __restrict__ Aptr, const float* __restrict__ W,
    const float* __restrict__ bias, void* __restrict__ Out, float oscale)
{
  __shared__ __align__(16) half_t Ash[128][40];   // [m][k], 80B rows (16B aligned)
  __shared__ __align__(16) half_t Bsh[128][40];   // [n][k] (transposed W tile)

  const int tid   = threadIdx.x;
  const int wave  = tid >> 5;
  const int lane  = tid & 31;
  const int hf    = lane >> 4;
  const int ln    = lane & 15;
  const int waveM = wave >> 1;      // 0..3
  const int waveN = wave & 1;       // 0..1
  const int m0 = blockIdx.y * 128;
  const int n0 = blockIdx.x * 128;

  v8f C[2][4] = {};

  const int arow = tid >> 1;          // 0..127
  const int akq  = (tid & 1) * 16;    // 0 / 16
  const int brow = tid >> 3;          // 0..31
  const int bnq  = (tid & 7) * 16;    // 0..112

  for (int kt = 0; kt < DM; kt += 32) {
    // --- stage A tile (128 x 32) as f16, b128 LDS stores ---
    {
      v8h h0, h1;
      if (A_F16) {
        const v8h* ap = (const v8h*)((const half_t*)Aptr + (size_t)(m0 + arow) * DM + kt + akq);
        h0 = ap[0]; h1 = ap[1];
      } else {
        const float4* ap = (const float4*)((const float*)Aptr + (size_t)(m0 + arow) * DM + kt + akq);
        float4 f0 = ap[0], f1 = ap[1], f2 = ap[2], f3 = ap[3];
        h0 = (v8h){(half_t)f0.x,(half_t)f0.y,(half_t)f0.z,(half_t)f0.w,
                   (half_t)f1.x,(half_t)f1.y,(half_t)f1.z,(half_t)f1.w};
        h1 = (v8h){(half_t)f2.x,(half_t)f2.y,(half_t)f2.z,(half_t)f2.w,
                   (half_t)f3.x,(half_t)f3.y,(half_t)f3.z,(half_t)f3.w};
      }
      *(v8h*)&Ash[arow][akq]     = h0;
      *(v8h*)&Ash[arow][akq + 8] = h1;
    }
    // --- stage W tile (32 x 128) transposed to [n][k] ---
    {
      const float4* bp = (const float4*)(W + (size_t)(kt + brow) * DM + n0 + bnq);
      float4 f0 = bp[0], f1 = bp[1], f2 = bp[2], f3 = bp[3];
      Bsh[bnq +  0][brow] = (half_t)f0.x; Bsh[bnq +  1][brow] = (half_t)f0.y;
      Bsh[bnq +  2][brow] = (half_t)f0.z; Bsh[bnq +  3][brow] = (half_t)f0.w;
      Bsh[bnq +  4][brow] = (half_t)f1.x; Bsh[bnq +  5][brow] = (half_t)f1.y;
      Bsh[bnq +  6][brow] = (half_t)f1.z; Bsh[bnq +  7][brow] = (half_t)f1.w;
      Bsh[bnq +  8][brow] = (half_t)f2.x; Bsh[bnq +  9][brow] = (half_t)f2.y;
      Bsh[bnq + 10][brow] = (half_t)f2.z; Bsh[bnq + 11][brow] = (half_t)f2.w;
      Bsh[bnq + 12][brow] = (half_t)f3.x; Bsh[bnq + 13][brow] = (half_t)f3.y;
      Bsh[bnq + 14][brow] = (half_t)f3.z; Bsh[bnq + 15][brow] = (half_t)f3.w;
    }
    if (kt + 32 < DM) {
      if (A_F16)
        __builtin_prefetch((const half_t*)Aptr + (size_t)(m0 + arow) * DM + kt + 32 + akq, 0, 1);
      else
        __builtin_prefetch((const float*)Aptr + (size_t)(m0 + arow) * DM + kt + 32 + akq, 0, 1);
      __builtin_prefetch(W + (size_t)(kt + 32 + brow) * DM + n0 + bnq, 0, 1);
    }
    __syncthreads();

    // --- gather fragments (2 x ds_load_b128 each) + WMMA ---
    FragH af[2];
    #pragma unroll
    for (int mt = 0; mt < 2; ++mt) {
      const uint4* arp = (const uint4*)&Ash[waveM * 32 + mt * 16 + ln][0];
      af[mt].q[0] = arp[hf];       // bytes 16*hf
      af[mt].q[1] = arp[2 + hf];   // bytes 32 + 16*hf
    }
    #pragma unroll
    for (int nt = 0; nt < 4; ++nt) {
      FragH bf;
      const uint4* brp = (const uint4*)&Bsh[waveN * 64 + nt * 16 + ln][0];
      bf.q[0] = brp[hf];           // bytes 16*hf
      bf.q[1] = brp[2 + hf];
      #pragma unroll
      for (int mt = 0; mt < 2; ++mt)
        C[mt][nt] = wmma_f16(af[mt].h, bf.h, C[mt][nt]);
    }
    __syncthreads();
  }

  // --- epilogue ---
  #pragma unroll
  for (int mt = 0; mt < 2; ++mt) {
    #pragma unroll
    for (int nt = 0; nt < 4; ++nt) {
      #pragma unroll
      for (int r = 0; r < 8; ++r) {
        const int m = m0 + waveM * 32 + mt * 16 + r + hf * 8;
        const int n = n0 + waveN * 64 + nt * 16 + ln;
        const float val = C[mt][nt][r] + bias[n];
        if (HEADMAJOR) {
          const int b = m >> 11;        // / TTQ
          const int t = m & (TTQ - 1);
          const int h = n >> 6;         // / DHH
          const int d = n & (DHH - 1);
          ((half_t*)Out)[(((size_t)(b * HH + h) * TTQ) + t) * DHH + d] = (half_t)(val * oscale);
        } else {
          ((float*)Out)[(size_t)m * DM + n] = val;
        }
      }
    }
  }
}

// ---------------------------------------------------------------------------
// Flash attention: per (t-tile of 64, h, b). 4 waves; each wave owns 16 Q rows.
// Q pre-scaled by 1/sqrt(DH). K chunk via async global->LDS DMA.
// Softmax denominator accumulated on the matrix core (P x ones WMMA);
// row max reduced on the VALU via DPP ROW_XMASK.
// ---------------------------------------------------------------------------
__global__ __launch_bounds__(128) void attn_flash_kernel(
    const half_t* __restrict__ Qw, const half_t* __restrict__ Kw,
    const half_t* __restrict__ Vw, const int* __restrict__ lens,
    half_t* __restrict__ Cw)
{
  __shared__ __align__(16) half_t Ksh[32][72];      // [s][d], 144B rows
  __shared__ __align__(16) half_t Vsh[64][40];      // [d][s] (transposed V chunk)
  __shared__ __align__(16) half_t Psh[4][16][40];   // per-wave P staging [m][s]

  const int tid  = threadIdx.x;
  const int wave = tid >> 5;
  const int lane = tid & 31;
  const int hf   = lane >> 4;
  const int ln   = lane & 15;
  const int b = blockIdx.z, h = blockIdx.y;
  const int tbase = blockIdx.x * 64 + wave * 16;
  const int len = lens[b];

  const half_t* Qbh = Qw + ((size_t)(b * HH + h) * TTQ + tbase) * DHH;
  const half_t* Kbh = Kw + (size_t)(b * HH + h) * TSS * DHH;
  const half_t* Vbh = Vw + (size_t)(b * HH + h) * TSS * DHH;

  // Q fragments (16 rows x 64 d), resident: 4x global_load_b128 per lane
  FragH qa[2];
  {
    const uint4* qrow = (const uint4*)(Qbh + (size_t)ln * DHH);
    #pragma unroll
    for (int ks = 0; ks < 2; ++ks) {
      qa[ks].q[0] = qrow[ks * 4 + hf];
      qa[ks].q[1] = qrow[ks * 4 + 2 + hf];
    }
  }

  // all-ones B fragment: rowsum(P) via one extra WMMA per chunk
  FragH ones;
  #pragma unroll
  for (int j = 0; j < 8; ++j) ones.u[j] = 0x3C003C00u;   // {1.0h, 1.0h}

  v8f O[4] = {};
  v8f Oext = {};                      // running softmax denominator per row
  float mrow[8];
  #pragma unroll
  for (int r = 0; r < 8; ++r) mrow[r] = -1e30f;

  const int srow = tid >> 2;         // 0..31
  const int d0   = (tid & 3) * 16;   // 0..48
  const unsigned klds0 = (unsigned)(size_t)&Ksh[srow][d0];
  const unsigned klds1 = (unsigned)(size_t)&Ksh[srow][d0 + 8];

  for (int s0 = 0; s0 < TSS; s0 += 32) {
    // --- K chunk via async DMA; V chunk manual (needs transpose) ---
    {
      const half_t* kp = Kbh + (size_t)(s0 + srow) * DHH + d0;
      async_load_b128(klds0, kp);
      async_load_b128(klds1, kp + 8);
      const half_t* vp = Vbh + (size_t)(s0 + srow) * DHH + d0;
      #pragma unroll
      for (int j = 0; j < 16; ++j) Vsh[d0 + j][srow] = vp[j];
      if (s0 + 32 < TSS) {
        __builtin_prefetch(kp + 32 * DHH, 0, 1);
        __builtin_prefetch(vp + 32 * DHH, 0, 1);
      }
    }
    wait_async0();
    __syncthreads();

    // --- S = Q * K^T : two 16x16 col tiles, 2 k-steps over d ---
    v8f c0 = {}, c1 = {};
    #pragma unroll
    for (int ks = 0; ks < 2; ++ks) {
      FragH kb0, kb1;
      const uint4* kr0 = (const uint4*)&Ksh[ln][0];
      const uint4* kr1 = (const uint4*)&Ksh[16 + ln][0];
      kb0.q[0] = kr0[ks * 4 + hf]; kb0.q[1] = kr0[ks * 4 + 2 + hf];
      kb1.q[0] = kr1[ks * 4 + hf]; kb1.q[1] = kr1[ks * 4 + 2 + hf];
      c0 = wmma_f16(qa[ks].h, kb0.h, c0);
      c1 = wmma_f16(qa[ks].h, kb1.h, c1);
    }

    // --- online softmax (rows m = r + hf*8, cols = ln / ln+16) ---
    const float msk0 = ((s0 + ln)      < len) ? 0.0f : -1e30f;
    const float msk1 = ((s0 + 16 + ln) < len) ? 0.0f : -1e30f;
    float alpha[8], p0[8], p1[8];
    #pragma unroll
    for (int r = 0; r < 8; ++r) {
      const float a0 = c0[r] + msk0;
      const float a1 = c1[r] + msk1;
      const float mx = rowmax16_dpp(fmaxf(a0, a1));
      const float mn = fmaxf(mrow[r], mx);
      alpha[r] = __expf(mrow[r] - mn);
      p0[r] = __expf(a0 - mn);
      p1[r] = __expf(a1 - mn);
      mrow[r] = mn;
    }
    #pragma unroll
    for (int r = 0; r < 8; ++r) {
      Oext[r] *= alpha[r];
      #pragma unroll
      for (int nd = 0; nd < 4; ++nd) O[nd][r] *= alpha[r];
    }

    // --- stage P (16x32) in per-wave LDS, regather as A-fragment ---
    #pragma unroll
    for (int r = 0; r < 8; ++r) {
      Psh[wave][r + hf * 8][ln]      = (half_t)p0[r];
      Psh[wave][r + hf * 8][ln + 16] = (half_t)p1[r];
    }
    FragH pf;
    {
      const uint4* prp = (const uint4*)&Psh[wave][ln][0];
      pf.q[0] = prp[hf];
      pf.q[1] = prp[2 + hf];
    }

    // --- O += P * V (4 n-tiles) and Oext += P * 1 (rowsum on matrix core) ---
    #pragma unroll
    for (int nd = 0; nd < 4; ++nd) {
      FragH bf;
      const uint4* vrp = (const uint4*)&Vsh[nd * 16 + ln][0];
      bf.q[0] = vrp[hf]; bf.q[1] = vrp[2 + hf];
      O[nd] = wmma_f16(pf.h, bf.h, O[nd]);
    }
    Oext = wmma_f16(pf.h, ones.h, Oext);
    __syncthreads();
  }

  // --- normalize and write ctx f16, token-major [B, T, H*DH] ---
  #pragma unroll
  for (int r = 0; r < 8; ++r) {
    const float inv = 1.0f / Oext[r];   // every lane holds rowsum for its rows
    const int m = r + hf * 8;
    half_t* crow = Cw + (size_t)(b * TTQ + tbase + m) * (HH * DHH) + h * DHH;
    #pragma unroll
    for (int nd = 0; nd < 4; ++nd)
      crow[nd * 16 + ln] = (half_t)(O[nd][r] * inv);
  }
}

extern "C" void kernel_launch(void* const* d_in, const int* in_sizes, int n_in,
                              void* d_out, int out_size, void* d_ws, size_t ws_size,
                              hipStream_t stream) {
  const float* x_Q = (const float*)d_in[0];
  const float* x_K = (const float*)d_in[1];
  const float* x_V = (const float*)d_in[2];
  const float* Wq  = (const float*)d_in[3];
  const float* bq  = (const float*)d_in[4];
  const float* Wk  = (const float*)d_in[5];
  const float* bk  = (const float*)d_in[6];
  const float* Wv  = (const float*)d_in[7];
  const float* bv  = (const float*)d_in[8];
  const float* Wo  = (const float*)d_in[9];
  const float* bo  = (const float*)d_in[10];
  const int*   lens = (const int*)d_in[11];

  const size_t QKV = (size_t)BB * HH * TTQ * DHH;
  half_t* Qw = (half_t*)d_ws;
  half_t* Kw = Qw + QKV;
  half_t* Vw = Kw + QKV;
  half_t* Cw = Vw + QKV;

  dim3 gblock(256);
  dim3 ggrid(DM / 128, MM / 128);   // (8, 64)

  // QKV projections -> f16 head-major; Q pre-scaled by 1/sqrt(DH)
  gemm_wmma_kernel<false, true><<<ggrid, gblock, 0, stream>>>((const void*)x_Q, Wq, bq, (void*)Qw, 0.125f);
  gemm_wmma_kernel<false, true><<<ggrid, gblock, 0, stream>>>((const void*)x_K, Wk, bk, (void*)Kw, 1.0f);
  gemm_wmma_kernel<false, true><<<ggrid, gblock, 0, stream>>>((const void*)x_V, Wv, bv, (void*)Vw, 1.0f);

  dim3 agrid(TTQ / 64, HH, BB);     // (32, 16, 4)
  attn_flash_kernel<<<agrid, dim3(128), 0, stream>>>(Qw, Kw, Vw, lens, Cw);

  gemm_wmma_kernel<true, false><<<ggrid, gblock, 0, stream>>>((const void*)Cw, Wo, bo, d_out, 1.0f);
}
